// Subject_RTM_SNN_35699768164601
// MI455X (gfx1250) — compile-verified, matrix operand order checked by reading, not run on hardware
//
#include <hip/hip_runtime.h>
#include <hip/hip_bf16.h>

// ---------------------------------------------------------------------------
// Types
// ---------------------------------------------------------------------------
typedef __bf16 bf16;
typedef __attribute__((ext_vector_type(2)))  float  v2f;
typedef __attribute__((ext_vector_type(2)))  __bf16 v2bf;
typedef __attribute__((ext_vector_type(8)))  __bf16 v8bf;
typedef __attribute__((ext_vector_type(16))) __bf16 v16bf;
typedef __attribute__((ext_vector_type(8)))  float  v8f;
typedef __attribute__((ext_vector_type(4)))  unsigned int v4u;
typedef __attribute__((ext_vector_type(4)))  int  v4i;
typedef __attribute__((ext_vector_type(8)))  int  v8i;

// Problem constants
#define BATCH 256
#define TSEQ  1000
#define FIN   310
#define H1    128
#define H2    64
#define MROWS (BATCH * TSEQ)          // 256000
#define NT1   10                      // K-steps for GEMM1 (310 -> 320 / 32)
#define NT2   4                       // K-steps for GEMM2 (128 / 32)

// Workspace layout (bytes)
#define OFF_X1    ((size_t)0)                     // f32 [M,128]  131,072,000
#define OFF_SPK1  ((size_t)131072000)             // bf16 [M,128]  65,536,000
#define OFF_X2    ((size_t)196608000)             // f32 [M,64]    65,536,000
#define OFF_SPK2  ((size_t)262144000)             // bf16 [M,64]   32,768,000
#define OFF_ALOG  ((size_t)294912000)             // f32 [M]        1,024,000
#define OFF_FEAT  ((size_t)295936000)             // f32 [256,64]      65,536
#define OFF_RATE  ((size_t)296001536)             // f32 [2]

// ---------------------------------------------------------------------------
// Tensor Data Mover: 2D tile load Global -> LDS (D# per ISA cdna5 §8).
// ---------------------------------------------------------------------------
__device__ __forceinline__ void tdm_load_2d(
    unsigned lds_addr, const void* gptr,
    unsigned data_size_code,           // 0=1B,1=2B,2=4B
    unsigned tensor_dim0, unsigned tensor_dim1,
    unsigned tile_dim0, unsigned tile_dim1, unsigned dim0_stride)
{
    size_t ga = (size_t)gptr;
    v4u g0 = { 1u, lds_addr, (unsigned)ga,
               (unsigned)((ga >> 32) & 0x1FFFFFFu) | 0x80000000u };
    v8i g1 = { (int)(data_size_code << 16),
               (int)(tensor_dim0 << 16),
               (int)((tensor_dim0 >> 16) | ((tensor_dim1 & 0xFFFFu) << 16)),
               (int)((tensor_dim1 >> 16) | (tile_dim0 << 16)),
               (int)tile_dim1,
               (int)dim0_stride,
               0, 0 };
    v4i gz = {};
#if defined(__clang_major__) && (__clang_major__ >= 23)
    v8i gz8 = {};
    __builtin_amdgcn_tensor_load_to_lds(g0, g1, gz, gz, gz8, 0);
#else
    __builtin_amdgcn_tensor_load_to_lds(g0, g1, gz, gz, 0);
#endif
}

// ---------------------------------------------------------------------------
// Fragment helpers: 16-bit A/B layout (ISA 7.12.2).
// ---------------------------------------------------------------------------
__device__ __forceinline__ v16bf ldfrag_bf16(const bf16* base, int sub) {
    v8bf lo = *reinterpret_cast<const v8bf*>(base + sub * 8);
    v8bf hi = *reinterpret_cast<const v8bf*>(base + 16 + sub * 8);
    return __builtin_shufflevector(lo, hi, 0,1,2,3,4,5,6,7,8,9,10,11,12,13,14,15);
}

__device__ __forceinline__ v16bf cvtfrag_f32(const float* rowp, int sub) {
    v8f lo = *reinterpret_cast<const v8f*>(rowp + sub * 8);
    v8f hi = *reinterpret_cast<const v8f*>(rowp + 16 + sub * 8);
    v8bf lo8 = __builtin_convertvector(lo, v8bf);
    v8bf hi8 = __builtin_convertvector(hi, v8bf);
    return __builtin_shufflevector(lo8, hi8, 0,1,2,3,4,5,6,7,8,9,10,11,12,13,14,15);
}

// Pack two floats -> one packed-bf16 dword store (v_cvt_pk_bf16_f32)
__device__ __forceinline__ void pack2(bf16* dst, float a, float b) {
    v2f f = { a, b };
    *reinterpret_cast<v2bf*>(dst) = __builtin_convertvector(f, v2bf);
}

__device__ __forceinline__ float gelu_tanh(float x) {
    float x3 = x * x * x;
    return 0.5f * x * (1.0f + tanhf(0.7978845608028654f * (x + 0.044715f * x3)));
}

// B-tile staging, two-phase (batched loads, then packed LDS stores)
__device__ __forceinline__ void stage_B1(const float* __restrict__ W1,
                                         bf16* Bb, int kk, int tid) {
    float e0[8], e1[8];
    #pragma unroll
    for (int it = 0; it < 8; ++it) {
        int p = tid + it * 256;            // 0..2047 pairs
        int n = p & 127, kp = p >> 7;      // kp: 0..15
        int k = kk + kp * 2;
        e0[it] = (k     < FIN) ? W1[(long)k * H1 + n]       : 0.0f;
        e1[it] = (k + 1 < FIN) ? W1[(long)(k + 1) * H1 + n] : 0.0f;
    }
    #pragma unroll
    for (int it = 0; it < 8; ++it) {
        int p = tid + it * 256;
        int n = p & 127, kp = p >> 7;
        pack2(&Bb[n * 32 + kp * 2], e0[it], e1[it]);
    }
}

__device__ __forceinline__ void stage_B2(const float* __restrict__ W2,
                                         bf16* Bb, int kk, int tid) {
    float e0[4], e1[4];
    #pragma unroll
    for (int it = 0; it < 4; ++it) {
        int p = tid + it * 256;            // 0..1023 pairs
        int n = p & 63, kp = p >> 6;
        int k = kk + kp * 2;
        e0[it] = W2[(long)k * H2 + n];
        e1[it] = W2[(long)(k + 1) * H2 + n];
    }
    #pragma unroll
    for (int it = 0; it < 4; ++it) {
        int p = tid + it * 256;
        int n = p & 63, kp = p >> 6;
        pack2(&Bb[n * 32 + kp * 2], e0[it], e1[it]);
    }
}

// ---------------------------------------------------------------------------
// Kernel 0: zero the two spike-rate accumulators
// ---------------------------------------------------------------------------
__global__ void k_init(float* rateAcc) {
    if (threadIdx.x < 2) rateAcc[threadIdx.x] = 0.0f;
}

// ---------------------------------------------------------------------------
// Kernel 1: GEMM1 (M x 310 @ 310 x 128, bf16 WMMA, f32 acc)
//   Double-buffered: TDM DMA of f32 A-tile i+1 and B-staging overlap the
//   WMMAs on tile i.  Fused +b1, gelu, LayerNorm(128) -> x1 f32.
// ---------------------------------------------------------------------------
__global__ __launch_bounds__(256) void k_gemm1(
    const float* __restrict__ X, const float* __restrict__ W1,
    const float* __restrict__ B1, const float* __restrict__ LS,
    const float* __restrict__ LB, float* __restrict__ OUT)
{
    __shared__ __align__(16) float AbufF[2][128 * 32];  // 2 x 16KB, TDM dest
    __shared__ __align__(16) bf16  Bbuf[2][128 * 32];   // 2 x  8KB

    const int tid  = threadIdx.x;
    const int wave = tid >> 5;
    const int lane = tid & 31;
    const int sub  = lane >> 4;
    const int l15  = lane & 15;
    const long m0  = (long)blockIdx.x * 128;

    v8f acc[8];
    #pragma unroll
    for (int i = 0; i < 8; ++i) acc[i] = v8f{};

    // Prologue: load tile 0 into buffer 0
    if (wave == 0) {
        tdm_load_2d((unsigned)(size_t)(void*)&AbufF[0][0], X + m0 * FIN,
                    2u, (unsigned)FIN, (unsigned)(MROWS - m0),
                    32u, 128u, (unsigned)FIN);
    }
    stage_B1(W1, &Bbuf[0][0], 0, tid);
    if (wave == 0) __builtin_amdgcn_s_wait_tensorcnt(0);
    __syncthreads();

    for (int i = 0; i < NT1; ++i) {
        const int sel = i & 1, nsel = sel ^ 1;
        // Kick off tile i+1 while computing tile i
        if (i + 1 < NT1) {
            int kkn = (i + 1) * 32;
            if (wave == 0) {
                tdm_load_2d((unsigned)(size_t)(void*)&AbufF[nsel][0],
                            X + m0 * FIN + kkn,
                            2u, (unsigned)(FIN - kkn), (unsigned)(MROWS - m0),
                            32u, 128u, (unsigned)FIN);
            }
            stage_B1(W1, &Bbuf[nsel][0], kkn, tid);
        }
        // Compute tile i
        v16bf aF = cvtfrag_f32(&AbufF[sel][(wave * 16 + l15) * 32], sub);
        v16bf bF[8];
        #pragma unroll
        for (int nt = 0; nt < 8; ++nt)
            bF[nt] = ldfrag_bf16(&Bbuf[sel][(nt * 16 + l15) * 32], sub);
        #pragma unroll
        for (int nt = 0; nt < 8; ++nt)
            acc[nt] = __builtin_amdgcn_wmma_f32_16x16x32_bf16(
                false, aF, false, bF[nt], (short)0, acc[nt], false, false);

        if (wave == 0) __builtin_amdgcn_s_wait_tensorcnt(0);
        __syncthreads();
    }

    // Epilogue: +bias, gelu, LayerNorm across 128 cols, store f32.
    float b1v[8], sV[8], bV[8];
    #pragma unroll
    for (int nt = 0; nt < 8; ++nt) {
        int col = nt * 16 + l15;
        b1v[nt] = B1[col]; sV[nt] = LS[col]; bV[nt] = LB[col];
    }
    const int rbase = wave * 16 + sub * 8;
    #pragma unroll
    for (int r = 0; r < 8; ++r) {
        float g[8];
        float s1 = 0.0f, s2 = 0.0f;
        #pragma unroll
        for (int nt = 0; nt < 8; ++nt) {
            float xv = acc[nt][r] + b1v[nt];
            float gv = gelu_tanh(xv);
            g[nt] = gv; s1 += gv; s2 += gv * gv;
        }
        #pragma unroll
        for (int m = 1; m <= 8; m <<= 1) {     // reduce within 16-lane half
            s1 += __shfl_xor(s1, m, 32);
            s2 += __shfl_xor(s2, m, 32);
        }
        float mean = s1 * (1.0f / 128.0f);
        float var  = s2 * (1.0f / 128.0f) - mean * mean;
        float rstd = rsqrtf(var + 1e-6f);
        long  gr   = m0 + rbase + r;
        #pragma unroll
        for (int nt = 0; nt < 8; ++nt)
            OUT[gr * H1 + nt * 16 + l15] = (g[nt] - mean) * rstd * sV[nt] + bV[nt];
    }
}

// ---------------------------------------------------------------------------
// Kernel 2/4: LIF scan, one thread per (b, h) chain; spikes -> bf16.
// Unrolled x4 to batch the (chain-independent) global loads; spike count
// reduced across the wave before one atomic per wave.
// ---------------------------------------------------------------------------
template <int H>
__global__ __launch_bounds__(256) void k_lif(
    const float* __restrict__ XIN, const float* __restrict__ DECAY,
    bf16* __restrict__ SPK, float* __restrict__ rateAcc, int rateIdx)
{
    int tid = blockIdx.x * 256 + threadIdx.x;
    int b = tid / H, h = tid % H;
    float d = 1.0f / (1.0f + expf(-DECAY[h]));
    float v = 0.0f, z = 0.0f, cnt = 0.0f;
    long base = (long)b * TSEQ * H + h;
    for (int t = 0; t < TSEQ; t += 4) {
        const float* xp = XIN + base + (long)t * H;
        bf16*        sp = SPK + base + (long)t * H;
        float x0 = xp[0], x1 = xp[H], x2 = xp[2 * H], x3 = xp[3 * H];
        v = v * d * (1.0f - z) + x0;
        z = (v - 0.5f > 0.0f) ? 1.0f : 0.0f; sp[0]     = (bf16)z; cnt += z;
        v = v * d * (1.0f - z) + x1;
        z = (v - 0.5f > 0.0f) ? 1.0f : 0.0f; sp[H]     = (bf16)z; cnt += z;
        v = v * d * (1.0f - z) + x2;
        z = (v - 0.5f > 0.0f) ? 1.0f : 0.0f; sp[2 * H] = (bf16)z; cnt += z;
        v = v * d * (1.0f - z) + x3;
        z = (v - 0.5f > 0.0f) ? 1.0f : 0.0f; sp[3 * H] = (bf16)z; cnt += z;
    }
    #pragma unroll
    for (int m = 1; m < 32; m <<= 1) cnt += __shfl_xor(cnt, m, 32);
    if ((threadIdx.x & 31) == 0) atomicAdd(&rateAcc[rateIdx], cnt);
}

// ---------------------------------------------------------------------------
// Kernel 3: GEMM2 (M x 128 @ 128 x 64, bf16 WMMA), double-buffered TDM.
//   Fused +b2, LayerNorm(64) -> x2.
// ---------------------------------------------------------------------------
__global__ __launch_bounds__(256) void k_gemm2(
    const bf16* __restrict__ A, const float* __restrict__ W2,
    const float* __restrict__ B2, const float* __restrict__ LS,
    const float* __restrict__ LB, float* __restrict__ OUT)
{
    __shared__ __align__(16) bf16 Abuf[2][128 * 32];   // 2 x 8KB, TDM dest
    __shared__ __align__(16) bf16 Bbuf[2][64 * 32];    // 2 x 4KB

    const int tid  = threadIdx.x;
    const int wave = tid >> 5;
    const int lane = tid & 31;
    const int sub  = lane >> 4;
    const int l15  = lane & 15;
    const long m0  = (long)blockIdx.x * 128;

    v8f acc[4];
    #pragma unroll
    for (int i = 0; i < 4; ++i) acc[i] = v8f{};

    if (wave == 0) {
        tdm_load_2d((unsigned)(size_t)(void*)&Abuf[0][0], A + m0 * H1,
                    1u, (unsigned)H1, (unsigned)(MROWS - m0),
                    32u, 128u, (unsigned)H1);
    }
    stage_B2(W2, &Bbuf[0][0], 0, tid);
    if (wave == 0) __builtin_amdgcn_s_wait_tensorcnt(0);
    __syncthreads();

    for (int i = 0; i < NT2; ++i) {
        const int sel = i & 1, nsel = sel ^ 1;
        if (i + 1 < NT2) {
            int kkn = (i + 1) * 32;
            if (wave == 0) {
                tdm_load_2d((unsigned)(size_t)(void*)&Abuf[nsel][0],
                            A + m0 * H1 + kkn,
                            1u, (unsigned)(H1 - kkn), (unsigned)(MROWS - m0),
                            32u, 128u, (unsigned)H1);
            }
            stage_B2(W2, &Bbuf[nsel][0], kkn, tid);
        }
        v16bf aF = ldfrag_bf16(&Abuf[sel][(wave * 16 + l15) * 32], sub);
        v16bf bF[4];
        #pragma unroll
        for (int nt = 0; nt < 4; ++nt)
            bF[nt] = ldfrag_bf16(&Bbuf[sel][(nt * 16 + l15) * 32], sub);
        #pragma unroll
        for (int nt = 0; nt < 4; ++nt)
            acc[nt] = __builtin_amdgcn_wmma_f32_16x16x32_bf16(
                false, aF, false, bF[nt], (short)0, acc[nt], false, false);

        if (wave == 0) __builtin_amdgcn_s_wait_tensorcnt(0);
        __syncthreads();
    }

    float b2v[4], sV[4], bV[4];
    #pragma unroll
    for (int nt = 0; nt < 4; ++nt) {
        int col = nt * 16 + l15;
        b2v[nt] = B2[col]; sV[nt] = LS[col]; bV[nt] = LB[col];
    }
    const int rbase = wave * 16 + sub * 8;
    #pragma unroll
    for (int r = 0; r < 8; ++r) {
        float g[4];
        float s1 = 0.0f, s2 = 0.0f;
        #pragma unroll
        for (int nt = 0; nt < 4; ++nt) {
            float xv = acc[nt][r] + b2v[nt];
            g[nt] = xv; s1 += xv; s2 += xv * xv;
        }
        #pragma unroll
        for (int m = 1; m <= 8; m <<= 1) {
            s1 += __shfl_xor(s1, m, 32);
            s2 += __shfl_xor(s2, m, 32);
        }
        float mean = s1 * (1.0f / 64.0f);
        float var  = s2 * (1.0f / 64.0f) - mean * mean;
        float rstd = rsqrtf(var + 1e-6f);
        long  gr   = m0 + rbase + r;
        #pragma unroll
        for (int nt = 0; nt < 4; ++nt)
            OUT[gr * H2 + nt * 16 + l15] = (g[nt] - mean) * rstd * sV[nt] + bV[nt];
    }
}

// ---------------------------------------------------------------------------
// Kernel 5: attention logits a[r] = spk2[r,:] . Wa + ba
// ---------------------------------------------------------------------------
__global__ __launch_bounds__(256) void k_attnlogit(
    const bf16* __restrict__ SPK2, const float* __restrict__ Wa,
    const float* __restrict__ ba, float* __restrict__ ALOG)
{
    long r = (long)blockIdx.x * 256 + threadIdx.x;
    const bf16* p = SPK2 + r * H2;
    float s = 0.0f;
    #pragma unroll
    for (int k = 0; k < H2; ++k) s += (float)p[k] * Wa[k];
    ALOG[r] = s + ba[0];
}

// ---------------------------------------------------------------------------
// Kernel 6: per-batch softmax over T and weighted feature sum -> feat[b,64]
// ---------------------------------------------------------------------------
__global__ __launch_bounds__(128) void k_softmax_feat(
    const float* __restrict__ ALOG, const bf16* __restrict__ SPK2,
    float* __restrict__ FEAT)
{
    __shared__ float red[128];
    __shared__ float wbuf[TSEQ];
    const int b = blockIdx.x, tid = threadIdx.x;
    const float* a = ALOG + (long)b * TSEQ;

    float mx = -1e30f;
    for (int t = tid; t < TSEQ; t += 128) mx = fmaxf(mx, a[t]);
    red[tid] = mx; __syncthreads();
    for (int s = 64; s > 0; s >>= 1) {
        if (tid < s) red[tid] = fmaxf(red[tid], red[tid + s]);
        __syncthreads();
    }
    mx = red[0]; __syncthreads();

    float sm = 0.0f;
    for (int t = tid; t < TSEQ; t += 128) {
        float w = expf(a[t] - mx);
        wbuf[t] = w; sm += w;
    }
    red[tid] = sm; __syncthreads();
    for (int s = 64; s > 0; s >>= 1) {
        if (tid < s) red[tid] += red[tid + s];
        __syncthreads();
    }
    float inv = 1.0f / red[0];
    __syncthreads();

    if (tid < H2) {
        const bf16* sp = SPK2 + (long)b * TSEQ * H2 + tid;
        float acc = 0.0f;
        for (int t = 0; t < TSEQ; ++t)
            acc += (float)sp[(long)t * H2] * wbuf[t];
        FEAT[b * H2 + tid] = acc * inv;
    }
}

// ---------------------------------------------------------------------------
// Kernel 7: head GEMM [256,64]@[64,4] + bo -> logits; + rate scalar
// ---------------------------------------------------------------------------
__global__ __launch_bounds__(256) void k_head(
    const float* __restrict__ FEAT, const float* __restrict__ Wo,
    const float* __restrict__ bo, const float* __restrict__ rateAcc,
    float* __restrict__ OUT)
{
    int b = threadIdx.x;
    const float* f = FEAT + b * H2;
    #pragma unroll
    for (int c = 0; c < 4; ++c) {
        float s = bo[c];
        for (int k = 0; k < H2; ++k) s += f[k] * Wo[k * 4 + c];
        OUT[b * 4 + c] = s;
    }
    if (b == 0) {
        float r1 = rateAcc[0] / (float)((long)MROWS * H1);
        float r2 = rateAcc[1] / (float)((long)MROWS * H2);
        OUT[BATCH * 4] = 0.5f * (r1 + r2);
    }
}

// ---------------------------------------------------------------------------
// Launch
// ---------------------------------------------------------------------------
extern "C" void kernel_launch(void* const* d_in, const int* in_sizes, int n_in,
                              void* d_out, int out_size, void* d_ws, size_t ws_size,
                              hipStream_t stream) {
    const float* x_seq  = (const float*)d_in[0];
    const float* W1     = (const float*)d_in[1];
    const float* b1     = (const float*)d_in[2];
    const float* ln1_s  = (const float*)d_in[3];
    const float* ln1_b  = (const float*)d_in[4];
    const float* decay1 = (const float*)d_in[5];
    const float* W2     = (const float*)d_in[6];
    const float* b2     = (const float*)d_in[7];
    const float* ln2_s  = (const float*)d_in[8];
    const float* ln2_b  = (const float*)d_in[9];
    const float* decay2 = (const float*)d_in[10];
    const float* Wa     = (const float*)d_in[11];
    const float* ba     = (const float*)d_in[12];
    const float* Wo     = (const float*)d_in[13];
    const float* bo     = (const float*)d_in[14];
    float* out = (float*)d_out;

    char* ws = (char*)d_ws;
    float* x1      = (float*)(ws + OFF_X1);
    bf16*  spk1    = (bf16*)(ws + OFF_SPK1);
    float* x2      = (float*)(ws + OFF_X2);
    bf16*  spk2    = (bf16*)(ws + OFF_SPK2);
    float* alog    = (float*)(ws + OFF_ALOG);
    float* feat    = (float*)(ws + OFF_FEAT);
    float* rateAcc = (float*)(ws + OFF_RATE);

    k_init<<<1, 32, 0, stream>>>(rateAcc);

    k_gemm1<<<MROWS / 128, 256, 0, stream>>>(x_seq, W1, b1, ln1_s, ln1_b, x1);

    k_lif<H1><<<(BATCH * H1) / 256, 256, 0, stream>>>(x1, decay1, spk1, rateAcc, 0);

    k_gemm2<<<MROWS / 128, 256, 0, stream>>>(spk1, W2, b2, ln2_s, ln2_b, x2);

    k_lif<H2><<<(BATCH * H2) / 256, 256, 0, stream>>>(x2, decay2, spk2, rateAcc, 1);

    k_attnlogit<<<MROWS / 256, 256, 0, stream>>>(spk2, Wa, ba, alog);

    k_softmax_feat<<<BATCH, 128, 0, stream>>>(alog, spk2, feat);

    k_head<<<1, BATCH, 0, stream>>>(feat, Wo, bo, rateAcc, out);
}
